// EnhancedStrategySuperposition_63926293233889
// MI455X (gfx1250) — compile-verified, matrix operand order checked by reading, not run on hardware
//
#include <hip/hip_runtime.h>

// ---------------------------------------------------------------------------
// B=8192, D=1024, E=32, A=128
// out[b,a] = sum_e w[b,e] * ( sum_d x[b,d]*W_strat[e,a,d] + b_strat[e,a] )
// w = softmax(x@W_att.T + b_att + adaptive_bias + gumbel(gumbel_u)), TAU=1
//
//  k0) split x       -> bf16 hi/lo in ws (exact, once)
//  k1) split W_strat -> bf16 hi/lo in ws (exact, once)
//  k2) gating: wave per row, lane per expert, shfl softmax -> ws
//  k3) fused GEMM, expert-split x4 for occupancy: 256 blocks x 8 waves.
//      Per block: 8 experts, K=64 chunks, async global->LDS staging,
//      ping-pong LDS, 3-product split-bf16 WMMA, fold w[b,e] per expert,
//      write deterministic partial.
//  k4) reduce partials + sum_e w[b,e]*b_strat[e,a] -> out
// ---------------------------------------------------------------------------

#define BN 8192
#define DN 1024
#define EN 32
#define AN 128
#define ESPLIT 4                 // expert splits (8 experts per block)
#define EPB (EN / ESPLIT)        // 8
#define KCH 64                   // k-chunk width (halves)
#define LSTR 72                  // LDS row stride in halves (144B = 9*16B)
#define NHALF (128 * LSTR)       // ushorts per LDS array (18432 B)
#define NIT (EPB * (DN / KCH))   // 8 * 16 = 128 chunk iterations per block

typedef __bf16 bf16_t;
typedef bf16_t v16bf __attribute__((ext_vector_type(16)));
typedef float  v8f   __attribute__((ext_vector_type(8)));
union FragU { v16bf v; uint4 u[2]; };

__device__ __forceinline__ unsigned short f2bf(float f) {
    unsigned u = __float_as_uint(f);
    unsigned r = u + 0x7FFFu + ((u >> 16) & 1u);    // RNE
    return (unsigned short)(r >> 16);
}
__device__ __forceinline__ float bf2f(unsigned short h) {
    return __uint_as_float(((unsigned)h) << 16);
}

// CDNA5 async copy: 16B global -> LDS per lane, tracked by ASYNCcnt.
__device__ __forceinline__ void async_b128(unsigned lds_addr, const void* gaddr) {
    asm volatile("global_load_async_to_lds_b128 %0, %1, off"
                 :: "v"(lds_addr), "v"(gaddr) : "memory");
}
__device__ __forceinline__ void wait_async0() {
    asm volatile("s_wait_asynccnt 0x0" ::: "memory");
}

// ---------------------------------------------------------------------------
__global__ __launch_bounds__(256)
void ess_split_kernel(const float4* __restrict__ src, ushort4* __restrict__ hi,
                      ushort4* __restrict__ lo, int n4) {
    int i = blockIdx.x * 256 + threadIdx.x;
    if (i >= n4) return;
    float4 v = src[i];
    ushort4 h, l;
    h.x = f2bf(v.x); l.x = f2bf(v.x - bf2f(h.x));
    h.y = f2bf(v.y); l.y = f2bf(v.y - bf2f(h.y));
    h.z = f2bf(v.z); l.z = f2bf(v.z - bf2f(h.z));
    h.w = f2bf(v.w); l.w = f2bf(v.w - bf2f(h.w));
    hi[i] = h; lo[i] = l;
}

// ---------------------------------------------------------------------------
__global__ __launch_bounds__(256)
void ess_gate_kernel(const float* __restrict__ x, const float* __restrict__ Watt,
                     const float* __restrict__ batt, const float* __restrict__ abias,
                     const float* __restrict__ gu, float* __restrict__ wout) {
    const int lane = threadIdx.x & 31;
    const int wv   = threadIdx.x >> 5;
    const int b    = blockIdx.x * 8 + wv;

    const float4* xr = (const float4*)(x + (size_t)b * DN);
    const float4* wr = (const float4*)(Watt + (size_t)lane * DN);

    float acc = 0.f;
    #pragma unroll 4
    for (int i = 0; i < DN / 4; ++i) {
        float4 a = xr[i];
        float4 w = wr[i];
        acc = fmaf(a.x, w.x, fmaf(a.y, w.y, fmaf(a.z, w.z, fmaf(a.w, w.w, acc))));
    }
    float z = acc + batt[lane] + abias[lane];
    float u = gu[(size_t)b * EN + lane];
    z += -logf(-logf(u + 1e-10f) + 1e-10f);      // gumbel, TAU==1

    float m = z;
    #pragma unroll
    for (int off = 16; off > 0; off >>= 1) m = fmaxf(m, __shfl_xor(m, off, 32));
    float p = expf(z - m);
    float s = p;
    #pragma unroll
    for (int off = 16; off > 0; off >>= 1) s += __shfl_xor(s, off, 32);
    wout[(size_t)b * EN + lane] = p / s;
}

// ---------------------------------------------------------------------------
// k3: expert-split fused GEMM. grid = 64 btiles * ESPLIT; 256 threads.
// Block tile 128 rows x 128 cols; wave owns 16x128 (8 C tiles).
// ---------------------------------------------------------------------------
__global__ __launch_bounds__(256)
void ess_moe_kernel(const unsigned short* __restrict__ xh_g,
                    const unsigned short* __restrict__ xl_g,
                    const unsigned short* __restrict__ wh_g,
                    const unsigned short* __restrict__ wl_g,
                    const float* __restrict__ wgt,
                    float* __restrict__ part) {
    extern __shared__ unsigned short smem[];   // 2 bufs x 4 arrays x NHALF

    const int tid   = threadIdx.x;
    const int lane  = tid & 31;
    const int wv    = tid >> 5;
    const int bt    = blockIdx.x & 63;
    const int es    = blockIdx.x >> 6;         // expert split id
    const int b0    = bt * 128;
    const int e0    = es * EPB;
    const int srow  = tid >> 1;                // staging row 0..127
    const int shalf = tid & 1;                 // staging half of 64-wide chunk
    const int rowA  = 16 * wv + (lane & 15);   // fragment row/col (ISA 7.12.2)
    const int koff  = (lane < 16) ? 0 : 8;

    v8f accO[8], accS[8];
    #pragma unroll
    for (int i = 0; i < 8; ++i) {
        accO[i] = (v8f){0.f,0.f,0.f,0.f,0.f,0.f,0.f,0.f};
        accS[i] = (v8f){0.f,0.f,0.f,0.f,0.f,0.f,0.f,0.f};
    }

    const size_t xoff = (size_t)(b0 + srow) * DN + shalf * 32;

    auto stage = [&](int it, int buf) {
        const int kc = it & 15;
        const int e  = e0 + (it >> 4);
        unsigned short* base = smem + buf * (4 * NHALF);
        unsigned lx = (unsigned)(unsigned long long)(base + srow * LSTR + shalf * 32);
        const unsigned short* gxh = xh_g + xoff + kc * KCH;
        const unsigned short* gxl = xl_g + xoff + kc * KCH;
        const size_t woffe = ((size_t)e * AN + srow) * DN + kc * KCH + shalf * 32;
        const unsigned short* gwh = wh_g + woffe;
        const unsigned short* gwl = wl_g + woffe;
        #pragma unroll
        for (int q = 0; q < 4; ++q) {
            async_b128(lx + 16 * q,             gxh + 8 * q);
            async_b128(lx + 2 * NHALF + 16 * q, gxl + 8 * q);
            async_b128(lx + 4 * NHALF + 16 * q, gwh + 8 * q);
            async_b128(lx + 6 * NHALF + 16 * q, gwl + 8 * q);
        }
    };

    stage(0, 0);

    for (int it = 0; it < NIT; ++it) {
        const int buf = it & 1;
        wait_async0();          // this wave's copies for chunk `it` landed
        __syncthreads();        // all waves' copies landed; prior reads retired
        if (it + 1 < NIT) stage(it + 1, buf ^ 1);   // overlap with WMMAs

        const unsigned short* xh = smem + buf * (4 * NHALF);
        const unsigned short* xl = xh + NHALF;
        const unsigned short* wh = xh + 2 * NHALF;
        const unsigned short* wl = xh + 3 * NHALF;

        #pragma unroll
        for (int s = 0; s < 2; ++s) {              // two 32-wide k sub-chunks
            const int kb = s * 32;
            FragU ah, al;
            ah.u[0] = *(const uint4*)&xh[rowA * LSTR + kb + koff];
            ah.u[1] = *(const uint4*)&xh[rowA * LSTR + kb + 16 + koff];
            al.u[0] = *(const uint4*)&xl[rowA * LSTR + kb + koff];
            al.u[1] = *(const uint4*)&xl[rowA * LSTR + kb + 16 + koff];

            #pragma unroll
            for (int ct = 0; ct < 8; ++ct) {
                const int acol = 16 * ct + (lane & 15);
                FragU bh, bl;
                bh.u[0] = *(const uint4*)&wh[acol * LSTR + kb + koff];
                bh.u[1] = *(const uint4*)&wh[acol * LSTR + kb + 16 + koff];
                bl.u[0] = *(const uint4*)&wl[acol * LSTR + kb + koff];
                bl.u[1] = *(const uint4*)&wl[acol * LSTR + kb + 16 + koff];
                // split-bf16 f32 emulation: hi*hi + hi*lo + lo*hi
                accS[ct] = __builtin_amdgcn_wmma_f32_16x16x32_bf16(
                    false, ah.v, false, bh.v, (short)0, accS[ct], false, false);
                accS[ct] = __builtin_amdgcn_wmma_f32_16x16x32_bf16(
                    false, ah.v, false, bl.v, (short)0, accS[ct], false, false);
                accS[ct] = __builtin_amdgcn_wmma_f32_16x16x32_bf16(
                    false, al.v, false, bh.v, (short)0, accS[ct], false, false);
            }
        }

        if ((it & 15) == 15) {                     // end of expert: fold w[b,e]
            const int e    = e0 + (it >> 4);
            const int rloc = b0 + 16 * wv + ((lane >> 4) << 3);
            float w8[8];
            #pragma unroll
            for (int v = 0; v < 8; ++v) w8[v] = wgt[(size_t)(rloc + v) * EN + e];
            #pragma unroll
            for (int ct = 0; ct < 8; ++ct)
                #pragma unroll
                for (int v = 0; v < 8; ++v) {
                    accO[ct][v] = fmaf(w8[v], accS[ct][v], accO[ct][v]);
                    accS[ct][v] = 0.f;
                }
        }
    }

    // store deterministic partial: part[es][row][col]
    float* pp = part + (size_t)es * BN * AN;
    const int rbase = b0 + 16 * wv + ((lane >> 4) << 3);
    const int cbase = lane & 15;
    #pragma unroll
    for (int ct = 0; ct < 8; ++ct) {
        const int col = 16 * ct + cbase;
        #pragma unroll
        for (int v = 0; v < 8; ++v)
            pp[(size_t)(rbase + v) * AN + col] = accO[ct][v];
    }
}

// ---------------------------------------------------------------------------
// k4: out = sum_s part[s] + sum_e w[b,e]*b_strat[e,a]
// ---------------------------------------------------------------------------
__global__ __launch_bounds__(256)
void ess_reduce_kernel(const float4* __restrict__ part,
                       const float* __restrict__ wgt,
                       const float4* __restrict__ bst4,
                       float4* __restrict__ out4) {
    const int i = blockIdx.x * 256 + threadIdx.x;       // float4 index
    const int n4 = BN * AN / 4;
    if (i >= n4) return;
    const size_t ps = (size_t)BN * AN / 4;
    float4 s = part[i];
    #pragma unroll
    for (int k = 1; k < ESPLIT; ++k) {
        float4 p = part[(size_t)k * ps + i];
        s.x += p.x; s.y += p.y; s.z += p.z; s.w += p.w;
    }
    const int row  = (i * 4) / AN;
    const int c4   = (i * 4 % AN) / 4;
    for (int e = 0; e < EN; ++e) {
        float  w = wgt[(size_t)row * EN + e];
        float4 b = bst4[e * (AN / 4) + c4];
        s.x = fmaf(w, b.x, s.x); s.y = fmaf(w, b.y, s.y);
        s.z = fmaf(w, b.z, s.z); s.w = fmaf(w, b.w, s.w);
    }
    out4[i] = s;
}

// ---------------------------------------------------------------------------
extern "C" void kernel_launch(void* const* d_in, const int* in_sizes, int n_in,
                              void* d_out, int out_size, void* d_ws, size_t ws_size,
                              hipStream_t stream) {
    const float* x     = (const float*)d_in[0];  // [B, D]
    const float* W_att = (const float*)d_in[1];  // [E, D]
    const float* b_att = (const float*)d_in[2];  // [E]
    const float* abias = (const float*)d_in[3];  // [E]
    const float* W_str = (const float*)d_in[4];  // [E, A, D]
    const float* b_str = (const float*)d_in[5];  // [E, A]
    const float* gumb  = (const float*)d_in[6];  // [B, E]
    float*       out   = (float*)d_out;          // [B, A]

    // ws: wgt 1MB | xh 16MB | xl 16MB | wh 8MB | wl 8MB | partials 16MB
    char* ws = (char*)d_ws;
    float*          wgt  = (float*)ws;                 size_t o = (size_t)BN * EN * 4;
    unsigned short* xh   = (unsigned short*)(ws + o);  o += (size_t)BN * DN * 2;
    unsigned short* xl   = (unsigned short*)(ws + o);  o += (size_t)BN * DN * 2;
    unsigned short* wh   = (unsigned short*)(ws + o);  o += (size_t)EN * AN * DN * 2;
    unsigned short* wl   = (unsigned short*)(ws + o);  o += (size_t)EN * AN * DN * 2;
    float*          part = (float*)(ws + o);

    const int nx4 = BN * DN / 4;          // 2,097,152
    const int nw4 = EN * AN * DN / 4;     // 1,048,576
    ess_split_kernel<<<nx4 / 256, 256, 0, stream>>>(
        (const float4*)x, (ushort4*)xh, (ushort4*)xl, nx4);
    ess_split_kernel<<<nw4 / 256, 256, 0, stream>>>(
        (const float4*)W_str, (ushort4*)wh, (ushort4*)wl, nw4);
    ess_gate_kernel<<<BN / 8, 256, 0, stream>>>(x, W_att, b_att, abias, gumb, wgt);

    const size_t lds_bytes = 2u * 4u * NHALF * sizeof(unsigned short);  // 147456
    ess_moe_kernel<<<64 * ESPLIT, 256, lds_bytes, stream>>>(
        xh, xl, wh, wl, wgt, part);

    ess_reduce_kernel<<<(BN * AN / 4) / 256, 256, 0, stream>>>(
        (const float4*)part, wgt, (const float4*)b_str, (float4*)out);
}
